// Block_47502338294589
// MI455X (gfx1250) — matrix operand order repaired and checked by reading, not compiled, not run on hardware
//
#include <hip/hip_runtime.h>
#include <hip/hip_bf16.h>

typedef __attribute__((ext_vector_type(16))) _Float16 v16h;
typedef __attribute__((ext_vector_type(8)))  float    v8f;

#define Tsz 1024
#define Csz 1024
#define Hn  16
#define Dh  64

__device__ __forceinline__ float gelu_tanh_f(float x){
  float x3 = x*x*x;
  return 0.5f*x*(1.f + tanhf(0.7978845608028654f*(x + 0.044715f*x3)));
}
__device__ __forceinline__ float gelu_erf_f(float x){
  return 0.5f*x*(1.f + erff(x*0.70710678118654752f));
}

// ---------------- LayerNorm (fp32 in) -> f16 out --------------------------
__global__ __launch_bounds__(256) void ln_f16_kernel(
    const float* __restrict__ x, const float* __restrict__ g,
    const float* __restrict__ b, _Float16* __restrict__ out)
{
  const int row = blockIdx.x;
  const float* xr = x + (size_t)row * Csz;
  float v[4]; float s = 0.f, s2 = 0.f;
  #pragma unroll
  for (int i = 0; i < 4; ++i){
    float t = xr[threadIdx.x + i*256];
    v[i] = t; s += t; s2 += t*t;
  }
  __shared__ float rs[256], rq[256];
  rs[threadIdx.x] = s; rq[threadIdx.x] = s2;
  __syncthreads();
  for (int off = 128; off > 0; off >>= 1){
    if (threadIdx.x < off){
      rs[threadIdx.x] += rs[threadIdx.x + off];
      rq[threadIdx.x] += rq[threadIdx.x + off];
    }
    __syncthreads();
  }
  const float mean = rs[0] * (1.f/Csz);
  const float var  = rq[0] * (1.f/Csz) - mean*mean;
  const float rstd = rsqrtf(var + 1e-5f);
  #pragma unroll
  for (int i = 0; i < 4; ++i){
    int idx = threadIdx.x + i*256;
    out[(size_t)row*Csz + idx] = (_Float16)((v[i]-mean)*rstd*g[idx] + b[idx]);
  }
}

// -------- transpose + convert: fp32 W[K][N] -> f16 Wt[N][K] ---------------
__global__ __launch_bounds__(256) void transpose_to_f16_kernel(
    const float* __restrict__ in, _Float16* __restrict__ out, int Kd, int Nd)
{
  __shared__ float tile[16][17];
  int n = blockIdx.x*16 + threadIdx.x;
  int k = blockIdx.y*16 + threadIdx.y;
  tile[threadIdx.y][threadIdx.x] = in[(size_t)k*Nd + n];
  __syncthreads();
  int n2 = blockIdx.x*16 + threadIdx.y;
  int k2 = blockIdx.y*16 + threadIdx.x;
  out[(size_t)n2*Kd + k2] = (_Float16)tile[threadIdx.x][threadIdx.y];
}

// ---------------- WMMA GEMM: out = act(A @ Bt^T + bias) [+ resid] ----------
// A: f16 [M][K] row-major; Bt: f16 [N][K] (pre-transposed weight).
// 256 threads = 8 waves (2 M x 4 N); block tile 64(M) x 256(N);
// wave tile 32x64 = 2x4 WMMA tiles -> 8 wmma per 12 ds_load_b128 per K-step.
// mode 0: normal stores (outf fp32 / outh f16). mode 1: qkv split stores.
__global__ __launch_bounds__(256) void gemm_wmma_kernel(
    const _Float16* __restrict__ A, const _Float16* __restrict__ Bt,
    const float* __restrict__ bias, const float* __restrict__ resid,
    float* __restrict__ outf, _Float16* __restrict__ outh,
    _Float16* __restrict__ qf, _Float16* __restrict__ kf, _Float16* __restrict__ vt,
    int M, int N, int K, int act, int mode)
{
  constexpr int LDT = 40;                 // padded f16 row stride (80B: conflict-free)
  __shared__ _Float16 sA[64*LDT];
  __shared__ _Float16 sB[256*LDT];
  const int tid  = threadIdx.x;
  const int lane = tid & 31;
  const int wave = tid >> 5;
  const int wm = wave & 1, wn = wave >> 1;       // 2 x 4 wave grid
  const int m0 = blockIdx.y*64, n0 = blockIdx.x*256;
  const int hlf = lane >> 4, l15 = lane & 15;

  v8f zero = {0.f,0.f,0.f,0.f,0.f,0.f,0.f,0.f};
  v8f acc[2][4];
  #pragma unroll
  for (int ms = 0; ms < 2; ++ms)
    #pragma unroll
    for (int ns = 0; ns < 4; ++ns) acc[ms][ns] = zero;

  for (int k0 = 0; k0 < K; k0 += 32){
    { int r = tid>>2, cc = tid&3;                // A tile: 64 rows x 4 chunks
      *(uint4*)(&sA[r*LDT + cc*8]) = *(const uint4*)(A + (size_t)(m0+r)*K + k0 + cc*8);
    }
    #pragma unroll
    for (int i = 0; i < 4; ++i){                 // B tile: 256 rows x 4 chunks
      int idx = tid + i*256, r = idx>>2, cc = idx&3;
      *(uint4*)(&sB[r*LDT + cc*8]) = *(const uint4*)(Bt + (size_t)(n0+r)*K + k0 + cc*8);
    }
    __syncthreads();
    v16h af[2], bf[4];
    #pragma unroll
    for (int ms = 0; ms < 2; ++ms){              // A frag: M=lane&15, K runs 8*hlf & 16+8*hlf
      int mr = wm*32 + ms*16 + l15;
      uint4* p = (uint4*)&af[ms];
      p[0] = *(const uint4*)(&sA[mr*LDT + 8*hlf]);
      p[1] = *(const uint4*)(&sA[mr*LDT + 16 + 8*hlf]);
    }
    #pragma unroll
    for (int ns = 0; ns < 4; ++ns){              // B frag: N=lane&15, K runs 8*hlf & 16+8*hlf
      int nr = wn*64 + ns*16 + l15;
      uint4* p = (uint4*)&bf[ns];
      p[0] = *(const uint4*)(&sB[nr*LDT + 8*hlf]);
      p[1] = *(const uint4*)(&sB[nr*LDT + 16 + 8*hlf]);
    }
    #pragma unroll
    for (int ms = 0; ms < 2; ++ms)
      #pragma unroll
      for (int ns = 0; ns < 4; ++ns)
        acc[ms][ns] = __builtin_amdgcn_wmma_f32_16x16x32_f16(
            false, af[ms], false, bf[ns], (short)0, acc[ms][ns], false, false);
    __syncthreads();
  }

  #pragma unroll
  for (int ms = 0; ms < 2; ++ms){
    #pragma unroll
    for (int ns = 0; ns < 4; ++ns){
      const int n = n0 + wn*64 + ns*16 + l15;
      const float bv = bias ? bias[n] : 0.f;
      #pragma unroll
      for (int r = 0; r < 8; ++r){
        const int m = m0 + wm*32 + ms*16 + r + 8*hlf;  // C layout: M = r + 8*half
        float val = acc[ms][ns][r] + bv;
        if (act == 1)      val = gelu_tanh_f(val);
        else if (act == 2) val = gelu_erf_f(val);
        if (mode == 0){
          if (resid) val += resid[(size_t)m*N + n];
          if (outf)  outf[(size_t)m*N + n] = val;
          if (outh)  outh[(size_t)m*N + n] = (_Float16)val;
        } else {
          const int bb = m >> 10, tt = m & (Tsz-1);
          if (n < Csz){
            const int hh = n >> 6, dd = n & 63;
            qf[(((size_t)(bb*Hn+hh))*Tsz + tt)*Dh + dd] = (_Float16)val;
          } else if (n < 2*Csz){
            const int n2 = n - Csz, hh = n2 >> 6, dd = n2 & 63;
            kf[(((size_t)(bb*Hn+hh))*Tsz + tt)*Dh + dd] = (_Float16)val;
          } else {
            const int n2 = n - 2*Csz, hh = n2 >> 6, dd = n2 & 63;
            vt[(((size_t)(bb*Hn+hh))*Dh + dd)*Tsz + tt] = (_Float16)val;  // V^T
          }
        }
      }
    }
  }
}

// ------------- causal flash attention, 1 wave / 16 queries -----------------
// q,k: f16 [b,h,t,d]; vT: f16 [b,h,d,t]; out: f16 [b,t,h*Dh+d]
__global__ __launch_bounds__(32) void attn_wmma_kernel(
    const _Float16* __restrict__ qf, const _Float16* __restrict__ kf,
    const _Float16* __restrict__ vt, _Float16* __restrict__ of)
{
  const int lane = threadIdx.x;
  const int hlf = lane >> 4, l15 = lane & 15;
  const int bh = blockIdx.y;
  const int b = bh >> 4, h = bh & 15;
  const int q0 = blockIdx.x * 16;

  const _Float16* qb = qf + (size_t)bh * Tsz * Dh;
  const _Float16* kb = kf + (size_t)bh * Tsz * Dh;
  const _Float16* vb = vt + (size_t)bh * Dh * Tsz;

  v16h aq[2];                                   // Q A-frags, two 32-wide d chunks
  {
    const int tq = q0 + l15;
    #pragma unroll
    for (int dc = 0; dc < 2; ++dc){
      const _Float16* base = qb + (size_t)tq*Dh + dc*32;
      uint4* p = (uint4*)&aq[dc];
      p[0] = *(const uint4*)(base + 8*hlf);
      p[1] = *(const uint4*)(base + 16 + 8*hlf);
    }
  }

  v8f zero = {0.f,0.f,0.f,0.f,0.f,0.f,0.f,0.f};
  v8f o[4]; o[0]=zero; o[1]=zero; o[2]=zero; o[3]=zero;
  float mrow[8], lrow[8];
  #pragma unroll
  for (int r = 0; r < 8; ++r){ mrow[r] = -3.0e38f; lrow[r] = 0.f; }

  __shared__ _Float16 sP[16*40];                // P tile relayout C-frag -> A-frag

  const int ntiles = (q0 + 16 + 31) >> 5;       // kv tiles of 32; jb <= q0 always
  for (int j = 0; j < ntiles; ++j){
    const int jb = j << 5;
    v8f s0 = zero, s1 = zero;
    #pragma unroll
    for (int dc = 0; dc < 2; ++dc){             // S = Q K^T, accumulate over d
      const _Float16* base0 = kb + (size_t)(jb + l15)*Dh + dc*32 + 8*hlf;
      v16h bk0; uint4* p0 = (uint4*)&bk0;
      p0[0] = *(const uint4*)base0; p0[1] = *(const uint4*)(base0 + 16);
      s0 = __builtin_amdgcn_wmma_f32_16x16x32_f16(false, aq[dc], false, bk0, (short)0, s0, false, false);
      const _Float16* base1 = kb + (size_t)(jb + 16 + l15)*Dh + dc*32 + 8*hlf;
      v16h bk1; uint4* p1 = (uint4*)&bk1;
      p1[0] = *(const uint4*)base1; p1[1] = *(const uint4*)(base1 + 16);
      s1 = __builtin_amdgcn_wmma_f32_16x16x32_f16(false, aq[dc], false, bk1, (short)0, s1, false, false);
    }
    const bool need_mask = (jb + 31 > q0);
    float mnew[8];
    #pragma unroll
    for (int r = 0; r < 8; ++r){
      float v0 = s0[r]*0.125f, v1 = s1[r]*0.125f;   // 1/sqrt(64)
      if (need_mask){
        const int tq = q0 + r + 8*hlf;
        if (jb + l15 > tq)      v0 = -3.0e38f;
        if (jb + 16 + l15 > tq) v1 = -3.0e38f;
      }
      s0[r] = v0; s1[r] = v1;
      float mm = fmaxf(v0, v1);                     // row-max over 16 lanes of half
      mm = fmaxf(mm, __shfl_xor(mm, 1, 32));
      mm = fmaxf(mm, __shfl_xor(mm, 2, 32));
      mm = fmaxf(mm, __shfl_xor(mm, 4, 32));
      mm = fmaxf(mm, __shfl_xor(mm, 8, 32));
      mnew[r] = fmaxf(mrow[r], mm);
    }
    #pragma unroll
    for (int r = 0; r < 8; ++r){
      const float a  = __expf(mrow[r] - mnew[r]);
      const float p0 = __expf(s0[r] - mnew[r]);
      const float p1 = __expf(s1[r] - mnew[r]);
      sP[(r + 8*hlf)*40 + l15]      = (_Float16)p0;
      sP[(r + 8*hlf)*40 + 16 + l15] = (_Float16)p1;
      float rsum = p0 + p1;
      rsum += __shfl_xor(rsum, 1, 32);
      rsum += __shfl_xor(rsum, 2, 32);
      rsum += __shfl_xor(rsum, 4, 32);
      rsum += __shfl_xor(rsum, 8, 32);
      lrow[r] = lrow[r]*a + rsum;
      mrow[r] = mnew[r];
      o[0][r] *= a; o[1][r] *= a; o[2][r] *= a; o[3][r] *= a;
    }
    __syncthreads();
    v16h ap;                                        // P A-frag from LDS
    { uint4* p = (uint4*)&ap;
      p[0] = *(const uint4*)(&sP[l15*40 + 8*hlf]);
      p[1] = *(const uint4*)(&sP[l15*40 + 16 + 8*hlf]);
    }
    #pragma unroll
    for (int dt = 0; dt < 4; ++dt){                 // O += P V, B-frag from V^T
      const _Float16* base = vb + (size_t)(dt*16 + l15)*Tsz + jb + 8*hlf;
      v16h bv; uint4* p = (uint4*)&bv;
      p[0] = *(const uint4*)base; p[1] = *(const uint4*)(base + 16);
      o[dt] = __builtin_amdgcn_wmma_f32_16x16x32_f16(false, ap, false, bv, (short)0, o[dt], false, false);
    }
    __syncthreads();
  }
  #pragma unroll
  for (int dt = 0; dt < 4; ++dt){
    #pragma unroll
    for (int r = 0; r < 8; ++r){
      const int t = q0 + r + 8*hlf;
      const int d = dt*16 + l15;
      of[((size_t)(b*Tsz + t))*Csz + h*Dh + d] = (_Float16)(o[dt][r] / lrow[r]);
    }
  }
}

// ---- cross-attention collapse: per-batch matvec (vec[B][Kd] @ W[Kd][Nd]) --
__global__ __launch_bounds__(256) void matvec_kernel(
    const float* __restrict__ vec, const float* __restrict__ W,
    const float* __restrict__ bias, float* __restrict__ out, int Kd, int Nd)
{
  const int n = blockIdx.x*256 + threadIdx.x;
  const int b = blockIdx.y;
  float acc = bias[n];
  for (int k = 0; k < Kd; ++k) acc += vec[(size_t)b*Kd + k] * W[(size_t)k*Nd + n];
  out[(size_t)b*Nd + n] = acc;
}

__global__ __launch_bounds__(256) void bcast_add_kernel(
    float* __restrict__ x, const float* __restrict__ ca)
{
  const size_t idx = (size_t)blockIdx.x*256 + threadIdx.x;
  const int b = (int)(idx >> 20);          // T*C = 2^20
  const int c = (int)(idx & (Csz-1));
  x[idx] += ca[(size_t)b*Csz + c];
}

extern "C" void kernel_launch(void* const* d_in, const int* in_sizes, int n_in,
                              void* d_out, int out_size, void* d_ws, size_t ws_size,
                              hipStream_t stream)
{
  (void)in_sizes; (void)n_in; (void)out_size; (void)ws_size;
  const float* x       = (const float*)d_in[0];
  const float* vf      = (const float*)d_in[1];
  const float* ln1_g   = (const float*)d_in[2];
  const float* ln1_b   = (const float*)d_in[3];
  const float* ln2_g   = (const float*)d_in[4];
  const float* ln2_b   = (const float*)d_in[5];
  const float* attn_w  = (const float*)d_in[6];
  const float* attn_b  = (const float*)d_in[7];
  const float* attn_pw = (const float*)d_in[8];
  const float* attn_pb = (const float*)d_in[9];
  // d_in[10..13] (q_w,q_b,k_w,k_b): algebraically unused — identical keys =>
  // softmax is exactly uniform regardless of q/k.
  const float* v_w     = (const float*)d_in[14];
  const float* v_b     = (const float*)d_in[15];
  const float* ca_pw   = (const float*)d_in[16];
  const float* ca_pb   = (const float*)d_in[17];
  const float* fc_w    = (const float*)d_in[18];
  const float* fc_b    = (const float*)d_in[19];
  const float* mproj_w = (const float*)d_in[20];
  const float* mproj_b = (const float*)d_in[21];
  const float* down_w  = (const float*)d_in[22];
  const float* down_b  = (const float*)d_in[23];
  const float* up_w    = (const float*)d_in[24];
  const float* up_b    = (const float*)d_in[25];
  float* out = (float*)d_out;

  char* ws = (char*)d_ws;
  size_t off = 0;
  auto take = [&](size_t bytes) -> char* {
    char* p = ws + off; off += (bytes + 255) & ~(size_t)255; return p;
  };
  _Float16* H16    = (_Float16*)take((size_t)4096*1024*2);
  _Float16* WATTN  = (_Float16*)take((size_t)3072*1024*2);
  _Float16* WPROJ  = (_Float16*)take((size_t)1024*1024*2);
  _Float16* WFC    = (_Float16*)take((size_t)4096*1024*2);
  _Float16* WMPROJ = (_Float16*)take((size_t)1024*4096*2);
  _Float16* WDOWN  = (_Float16*)take((size_t)256*1024*2);
  _Float16* WUP    = (_Float16*)take((size_t)1024*256*2);
  _Float16* QF     = (_Float16*)take((size_t)4096*1024*2);
  _Float16* KF     = (_Float16*)take((size_t)4096*1024*2);
  _Float16* VT     = (_Float16*)take((size_t)4096*1024*2);
  _Float16* OF     = (_Float16*)take((size_t)4096*1024*2);
  float*    X1     = (float*)take((size_t)4096*1024*4);
  float*    X3     = (float*)take((size_t)4096*1024*4);
  _Float16* X3H    = (_Float16*)take((size_t)4096*1024*2);
  _Float16* AH     = (_Float16*)take((size_t)4096*256*2);
  float*    CAV    = (float*)take((size_t)4*1024*4);
  float*    CAO    = (float*)take((size_t)4*1024*4);
  _Float16* ACT    = QF;   // alias the 32MB QF/KF/VT/OF region (dead after proj GEMM)

  dim3 blk256(256), blk16x16(16,16), blk32(32);

  // weight prep: fp32 [K][N] -> f16 [N][K] (both GEMM operands become K-major)
  transpose_to_f16_kernel<<<dim3(3072/16, 1024/16), blk16x16, 0, stream>>>(attn_w,  WATTN,  1024, 3072);
  transpose_to_f16_kernel<<<dim3(1024/16, 1024/16), blk16x16, 0, stream>>>(attn_pw, WPROJ,  1024, 1024);
  transpose_to_f16_kernel<<<dim3(4096/16, 1024/16), blk16x16, 0, stream>>>(fc_w,    WFC,    1024, 4096);
  transpose_to_f16_kernel<<<dim3(1024/16, 4096/16), blk16x16, 0, stream>>>(mproj_w, WMPROJ, 4096, 1024);
  transpose_to_f16_kernel<<<dim3(256/16,  1024/16), blk16x16, 0, stream>>>(down_w,  WDOWN,  1024, 256);
  transpose_to_f16_kernel<<<dim3(1024/16, 256/16),  blk16x16, 0, stream>>>(up_w,    WUP,    256,  1024);

  // ln1 -> f16
  ln_f16_kernel<<<4096, blk256, 0, stream>>>(x, ln1_g, ln1_b, H16);

  // qkv GEMM (bias fused; writes q/k f16 [b,h,t,d] and V^T f16 [b,h,d,t])
  gemm_wmma_kernel<<<dim3(3072/256, 4096/64), blk256, 0, stream>>>(
      H16, WATTN, attn_b, nullptr, nullptr, nullptr, QF, KF, VT, 4096, 3072, 1024, 0, 1);

  // causal flash attention (WMMA for both S=QK^T and O=PV)
  attn_wmma_kernel<<<dim3(Tsz/16, 64), blk32, 0, stream>>>(QF, KF, VT, OF);

  // attn out-proj + residual(x) -> X1
  gemm_wmma_kernel<<<dim3(1024/256, 4096/64), blk256, 0, stream>>>(
      OF, WPROJ, attn_pb, x, X1, nullptr, nullptr, nullptr, nullptr, 4096, 1024, 1024, 0, 0);

  // cross-attention collapse: o = (vf@v_w+v_b)@ca_proj_w + ca_proj_b, broadcast over T
  matvec_kernel<<<dim3(1024/256, 4), blk256, 0, stream>>>(vf,  v_w,   v_b,   CAV, 1024, 1024);
  matvec_kernel<<<dim3(1024/256, 4), blk256, 0, stream>>>(CAV, ca_pw, ca_pb, CAO, 1024, 1024);
  bcast_add_kernel<<<dim3((4*1024*1024)/256), blk256, 0, stream>>>(X1, CAO);

  // ln2 -> f16 (reuse H16)
  ln_f16_kernel<<<4096, blk256, 0, stream>>>(X1, ln2_g, ln2_b, H16);

  // fc + tanh-GELU -> f16 activations
  gemm_wmma_kernel<<<dim3(4096/256, 4096/64), blk256, 0, stream>>>(
      H16, WFC, fc_b, nullptr, nullptr, ACT, nullptr, nullptr, nullptr, 4096, 4096, 1024, 1, 0);

  // mproj + residual(X1) -> X3 fp32 (+ f16 copy for adapter)
  gemm_wmma_kernel<<<dim3(1024/256, 4096/64), blk256, 0, stream>>>(
      ACT, WMPROJ, mproj_b, X1, X3, X3H, nullptr, nullptr, nullptr, 4096, 1024, 4096, 0, 0);

  // adapter down + exact GELU -> f16
  gemm_wmma_kernel<<<dim3(256/256, 4096/64), blk256, 0, stream>>>(
      X3H, WDOWN, down_b, nullptr, nullptr, AH, nullptr, nullptr, nullptr, 4096, 256, 1024, 2, 0);

  // adapter up + residual(X3) -> final output
  gemm_wmma_kernel<<<dim3(1024/256, 4096/64), blk256, 0, stream>>>(
      AH, WUP, up_b, X3, out, nullptr, nullptr, nullptr, nullptr, 4096, 1024, 256, 0, 0);
}
